// Self_Attention_4398046511602
// MI455X (gfx1250) — matrix operand order repaired and checked by reading, not compiled, not run on hardware
//
#include <hip/hip_runtime.h>
#include <hip/hip_bf16.h>

typedef __attribute__((ext_vector_type(16))) _Float16 v16h;
typedef __attribute__((ext_vector_type(8)))  _Float16 v8h;
typedef __attribute__((ext_vector_type(8)))  float    v8f;

#define B_ 4
#define S_ 4096
#define D_ 64

// Assemble a 16-bit A-fragment from two 16-byte halves (lo -> elements 0..7,
// hi -> elements 8..15; caller sets K order: table for Q/K, 16h+s for V^T).
__device__ __forceinline__ v16h make_frag(const _Float16* lo_p,
                                          const _Float16* hi_p) {
  v8h lo = *(const v8h*)lo_p;
  v8h hi = *(const v8h*)hi_p;
  v16h a;
#pragma unroll
  for (int e = 0; e < 8; ++e) { a[e] = lo[e]; a[8 + e] = hi[e]; }
  return a;
}

// ---------------------------------------------------------------------------
// Pass 1: x + pos_embed, then QKV projection.  Q (pre-scaled by 1/sqrt(64)),
// K stored row-major f16; V stored transposed (d-major) f16 so the attention
// kernel's V^T A-fragments are contiguous loads per lane.
// grid: B*S/4 blocks of 256 threads; each block handles 4 sequence rows.
// ---------------------------------------------------------------------------
__global__ __launch_bounds__(256) void qkv_kernel(
    const float* __restrict__ x, const float* __restrict__ w,
    _Float16* __restrict__ q, _Float16* __restrict__ k,
    _Float16* __restrict__ vt)
{
  __shared__ float wl[3 * 64 * 64];   // 48 KB of the 320 KB WGP LDS
  __shared__ float xe[4][64];

  const int tid = threadIdx.x;
  for (int i = tid; i < 3 * 64 * 64; i += 256) wl[i] = w[i];

  const int row = tid >> 6;                 // 0..3 (seq row within block)
  const int o   = tid & 63;                 // 0..63 (output / d column)
  const int bs  = blockIdx.x * 4 + row;     // flat (b,s)
  const int s   = bs & (S_ - 1);

  // pos_embed(s, d) = sin(s * 10000^(-2d/64)); lane loads x[bs][o], d == o
  float invf = __expf(-(2.0f * (float)o / 64.0f) * 9.210340371976184f);
  xe[row][o] = x[(size_t)bs * 64 + o] + __sinf((float)s * invf);
  __syncthreads();

  float accq = 0.f, acck = 0.f, accv = 0.f;
#pragma unroll
  for (int d = 0; d < 64; ++d) {
    float xv = xe[row][d];
    accq = fmaf(xv, wl[0 * 4096 + d * 64 + o], accq);
    acck = fmaf(xv, wl[1 * 4096 + d * 64 + o], acck);
    accv = fmaf(xv, wl[2 * 4096 + d * 64 + o], accv);
  }

  const size_t rowbase = (size_t)bs * 64;
  q[rowbase + o] = (_Float16)(accq * 0.125f);   // fold 1/sqrt(64) into Q
  k[rowbase + o] = (_Float16)acck;
  const int b = bs >> 12;                       // / S_
  vt[(size_t)b * 64 * S_ + (size_t)o * S_ + s] = (_Float16)accv;
}

// ---------------------------------------------------------------------------
// Pass 2: transposed flash-attention, one wave (wave32) per 16-query tile.
//   S^T = K x Q^T   (C layout: VGPR r = key r+8*half, lane = query)
//   softmax over keys = in-register tree + ONE cross-half shfl_xor(16)
//   O^T += V^T x P^T (P^T B-frag built from scores with 8 shfl_xor(16))
// K fragments are software-pipelined one key-tile ahead; V fragments are
// issued at iteration top and consumed after softmax -> score WMMAs never
// wait on fresh loads.  No LDS, no score materialization.
// block = 256 thr (8 waves), grid = B * S/128.
// ---------------------------------------------------------------------------
__global__ __launch_bounds__(256) void attn_kernel(
    const _Float16* __restrict__ q, const _Float16* __restrict__ k,
    const _Float16* __restrict__ vt, float* __restrict__ out)
{
  const int lane = threadIdx.x & 31;
  const int wave = threadIdx.x >> 5;
  const int half = lane >> 4;               // 0: lanes 0-15, 1: lanes 16-31
  const int l16  = lane & 15;
  const int b    = blockIdx.x >> 5;                           // / (S/128)
  const int q0   = ((blockIdx.x & 31) * 8 + wave) * 16;       // query base

  const _Float16* qb = q  + (size_t)b * S_ * 64;
  const _Float16* kb = k  + (size_t)b * S_ * 64;
  const _Float16* vb = vt + (size_t)b * 64 * S_;

  // Q^T B-fragments (loaded once): lane = query column l16,
  // K-dim rows d = 32c + {0..7}+8*half, 32c + {16..23}+8*half (table order).
  v16h bq[2];
  {
    const _Float16* qrow = qb + (size_t)(q0 + l16) * 64;
#pragma unroll
    for (int c = 0; c < 2; ++c)
      bq[c] = make_frag(qrow + 32 * c + 8 * half, qrow + 32 * c + 16 + 8 * half);
  }

  float m = -1e30f, l = 0.f;                // per-query (per-lane) stats
  v8f o0 = {}, o1 = {}, o2 = {}, o3 = {};   // O^T: VGPR r = d (16n+r+8h)

  // ---- prologue: K A-fragments for tile kt = 0
  v16h ak[2][2];   // [key subtile j][d chunk cd]
#pragma unroll
  for (int j = 0; j < 2; ++j)
#pragma unroll
    for (int cd = 0; cd < 2; ++cd) {
      const _Float16* kr = kb + (size_t)(16 * j + l16) * 64 + 32 * cd;
      ak[j][cd] = make_frag(kr + 8 * half, kr + 16 + 8 * half);
    }

  for (int kt = 0; kt < S_; kt += 32) {
    // ---- V^T A-frags for CURRENT tile: lane = d row 16n+l16
    v16h av[4];
#pragma unroll
    for (int n = 0; n < 4; ++n) {
      const _Float16* vr = vb + (size_t)(16 * n + l16) * S_ + kt;
      av[n] = make_frag(vr + 16 * half, vr + 16 * half + 8);
    }

    // ---- K A-frags for NEXT tile (clamped address keeps loop branch-free)
    const int ktn = (kt + 32 < S_) ? kt + 32 : 0;
    v16h akn[2][2];
#pragma unroll
    for (int j = 0; j < 2; ++j)
#pragma unroll
      for (int cd = 0; cd < 2; ++cd) {
        const _Float16* kr = kb + (size_t)(ktn + 16 * j + l16) * 64 + 32 * cd;
        akn[j][cd] = make_frag(kr + 8 * half, kr + 16 + 8 * half);
      }
    __builtin_prefetch(kb + (size_t)((kt + 64 < S_) ? kt + 64 : 0) * 64, 0, 0);

    // ---- S^T = K x Q^T, two 16-key subtiles (Q pre-scaled by 1/sqrt(64));
    // consumes fragments loaded one iteration ago -> no load wait here.
    v8f c01[2];
#pragma unroll
    for (int j = 0; j < 2; ++j) {
      v8f c = {};
      c = __builtin_amdgcn_wmma_f32_16x16x32_f16(
          false, ak[j][0], false, bq[0], (short)0, c, false, false);
      c = __builtin_amdgcn_wmma_f32_16x16x32_f16(
          false, ak[j][1], false, bq[1], (short)0, c, false, false);
      c01[j] = c;
    }

    // ---- online softmax over keys: register tree + one cross-half xchg
    float mloc = -1e30f;
#pragma unroll
    for (int r = 0; r < 8; ++r)
      mloc = fmaxf(mloc, fmaxf(c01[0][r], c01[1][r]));
    mloc = fmaxf(mloc, __shfl_xor(mloc, 16, 32));
    const float mnew  = fmaxf(m, mloc);
    const float scale = __expf(m - mnew);

    float p0[8], p1[8], rs = 0.f;
#pragma unroll
    for (int r = 0; r < 8; ++r) {
      p0[r] = __expf(c01[0][r] - mnew);
      p1[r] = __expf(c01[1][r] - mnew);
      rs += p0[r] + p1[r];
    }
    rs += __shfl_xor(rs, 16, 32);
    l = l * scale + rs;
    m = mnew;
#pragma unroll
    for (int r = 0; r < 8; ++r) {
      o0[r] *= scale; o1[r] *= scale; o2[r] *= scale; o3[r] *= scale;
    }

    // ---- build P^T B-fragment (lane = query col, K = 32 keys).
    // Partner half supplies the missing 8 keys: one shfl_xor(16) per r.
    v16h bp;
#pragma unroll
    for (int r = 0; r < 8; ++r) {
      float t  = half ? p0[r] : p1[r];
      float px = __shfl_xor(t, 16, 32);
      float lo = half ? px : p0[r];         // keys 16h + r
      float hi = half ? p1[r] : px;         // keys 16h + 8 + r
      bp[r]     = (_Float16)lo;
      bp[8 + r] = (_Float16)hi;
    }

    // ---- O^T += V^T x P^T over four 16-wide d chunks (av issued at top)
    o0 = __builtin_amdgcn_wmma_f32_16x16x32_f16(
        false, av[0], false, bp, (short)0, o0, false, false);
    o1 = __builtin_amdgcn_wmma_f32_16x16x32_f16(
        false, av[1], false, bp, (short)0, o1, false, false);
    o2 = __builtin_amdgcn_wmma_f32_16x16x32_f16(
        false, av[2], false, bp, (short)0, o2, false, false);
    o3 = __builtin_amdgcn_wmma_f32_16x16x32_f16(
        false, av[3], false, bp, (short)0, o3, false, false);

    // ---- rotate pipeline
#pragma unroll
    for (int j = 0; j < 2; ++j)
#pragma unroll
      for (int cd = 0; cd < 2; ++cd) ak[j][cd] = akn[j][cd];
  }

  // ---- epilogue: out[b][q0+l16][16n + 8h + r] = o[n][r] / l
  const float inv = 1.0f / l;
  float* orow = out + ((size_t)b * S_ + q0 + l16) * 64 + 8 * half;
#pragma unroll
  for (int n = 0; n < 4; ++n) {
    const v8f* oc = (n == 0) ? &o0 : (n == 1) ? &o1 : (n == 2) ? &o2 : &o3;
#pragma unroll
    for (int r = 0; r < 8; ++r)
      orow[16 * n + r] = (*oc)[r] * inv;    // 8 contiguous f32 -> b128 pairs
  }
}

extern "C" void kernel_launch(void* const* d_in, const int* in_sizes, int n_in,
                              void* d_out, int out_size, void* d_ws, size_t ws_size,
                              hipStream_t stream) {
  (void)in_sizes; (void)n_in; (void)out_size; (void)ws_size;
  const float* x = (const float*)d_in[0];     // [4,4096,64] f32
  const float* w = (const float*)d_in[1];     // [3,64,64]   f32
  float* out = (float*)d_out;                 // [4,4096,64] f32

  _Float16* qws = (_Float16*)d_ws;                       // 2 MB
  _Float16* kws = qws + (size_t)B_ * S_ * 64;            // 2 MB
  _Float16* vws = kws + (size_t)B_ * S_ * 64;            // 2 MB (d-major)

  qkv_kernel<<<B_ * S_ / 4, 256, 0, stream>>>(x, w, qws, kws, vws);
  attn_kernel<<<B_ * (S_ / 128), 256, 0, stream>>>(qws, kws, vws, out);
}
